// DenseEdgeConv_51943334477852
// MI455X (gfx1250) — matrix operand order repaired and checked by reading, not compiled, not run on hardware
//
#include <hip/hip_runtime.h>
#include <hip/hip_bf16.h>
#include <cstddef>

// ---------------------------------------------------------------------------
// DenseEdgeConv for MI455X (gfx1250): f16 WMMA 16x16x32, weights register-
// resident, async global->LDS gather pipeline (ASYNCcnt).
// B=4, C=64, N=8192, K=16, G=64, 3 layers.
// ---------------------------------------------------------------------------

#define NPTS 8192
#define NB   4
#define NCH  64
#define KNB  16
#define YSTR 264   // 256 channels + 8 halfs pad -> 528B row stride (bank-conflict-free b128)

typedef _Float16 v16h  __attribute__((ext_vector_type(16)));
typedef float    v8f   __attribute__((ext_vector_type(8)));
typedef unsigned int u4 __attribute__((ext_vector_type(4)));

union Frag {  // 32 bytes: one WMMA f16 A/B operand (8 VGPRs)
    v16h v;
    u4   u[2];
};

// ---------------------------------------------------------------------------
// CDNA5 async global->LDS copy (ASYNCcnt path).  vdst = LDS byte address
// (low 32 bits of the generic pointer == wave-relative LDS address per ISA
// flat-aperture mapping), vaddr = 64-bit global address.
// ---------------------------------------------------------------------------
__device__ __forceinline__ void async_ld_b128(unsigned ldsOff, const void* gaddr)
{
    asm volatile("global_load_async_to_lds_b128 %0, %1, off"
                 :: "v"(ldsOff), "v"(gaddr) : "memory");
}

template<int NOUT>
__device__ __forceinline__ void wait_async()
{
    asm volatile("s_wait_asynccnt %0" :: "n"(NOUT) : "memory");
}

// ---------------------------------------------------------------------------
// Kernel 1: transpose + convert x (B,C,N) f32 -> xT (B,N,C) f16
// ---------------------------------------------------------------------------
__global__ __launch_bounds__(256) void transpose_convert(const float* __restrict__ x,
                                                         _Float16* __restrict__ xT)
{
    __shared__ _Float16 sT[64][72];
    const int b   = blockIdx.y;
    const int n0  = blockIdx.x * 64;
    const int col = threadIdx.x & 63;   // n within tile
    const int r4  = threadIdx.x >> 6;   // 0..3

#pragma unroll
    for (int i = 0; i < 16; ++i) {
        const int c = r4 + i * 4;
        sT[col][c] = (_Float16)x[((size_t)b * NCH + c) * NPTS + n0 + col];
    }
    __syncthreads();
#pragma unroll
    for (int i = 0; i < 16; ++i) {
        const int n = r4 + i * 4;
        xT[((size_t)b * NPTS + n0 + n) * NCH + col] = sT[n][col];
    }
}

// ---------------------------------------------------------------------------
// Kernel 2: convert weights f32 -> f16.  W0 is FOLDED so that the gather can
// store raw [center ; knn] instead of [center ; knn-center]:
//   W0'[:,0:64] = W0[:,0:64] - W0[:,64:128],  W0'[:,64:128] = W0[:,64:128]
// since W0*[c; k-c] = (W0a-W0b)*c + W0b*k.
// ---------------------------------------------------------------------------
__global__ __launch_bounds__(256) void convert_weights(const float* __restrict__ W0,
                                                       const float* __restrict__ W1,
                                                       const float* __restrict__ W2,
                                                       _Float16* __restrict__ w0,
                                                       _Float16* __restrict__ w1,
                                                       _Float16* __restrict__ w2)
{
    const int i = blockIdx.x * 256 + threadIdx.x;
    if (i < 8192) {
        const int o = i >> 7, c = i & 127;
        float v = W0[i];
        if (c < 64) v -= W0[o * 128 + 64 + c];
        w0[i] = (_Float16)v;
    } else if (i < 16384) {
        w1[i - 8192] = (_Float16)W1[i - 8192];
    } else if (i < 28672) {
        w2[i - 16384] = (_Float16)W2[i - 16384];
    }
}

// ---------------------------------------------------------------------------
// Preload one layer's A-fragments (weights) into registers.
// CDNA5 A 16x32 layout: lane (lm,hi), row M=mt*16+lm, K = kt*32 + hi*8 + [0..7]
// (VGPRs 0-3) and +16 (VGPRs 4-7).
// ---------------------------------------------------------------------------
template<int CIN>
__device__ __forceinline__ void load_wfrags(const _Float16* __restrict__ wH,
                                            int lm, int hi, Frag wf[4][CIN / 32])
{
    constexpr int KT = CIN / 32;
#pragma unroll
    for (int mt = 0; mt < 4; ++mt) {
        const _Float16* arow = wH + (mt * 16 + lm) * CIN + hi * 8;
#pragma unroll
        for (int kt = 0; kt < KT; ++kt) {
            wf[mt][kt].u[0] = *(const u4*)(arow + kt * 32);
            wf[mt][kt].u[1] = *(const u4*)(arow + kt * 32 + 16);
        }
    }
}

// ---------------------------------------------------------------------------
// Async prefetch of one point's Y tile rows 0..127: [center(0..63); knn(64..127)]
// Each lane: k = lm neighbor row, cb = hi*32 channel chunk -> 8x b128 transfers.
// ---------------------------------------------------------------------------
__device__ __forceinline__ void prefetch_point(const _Float16* __restrict__ xT,
                                               const int* __restrict__ idx,
                                               int b, int n, int k, int cb,
                                               _Float16* dstRow)
{
    const int m = idx[((size_t)b * NPTS + n) * KNB + k];
    const _Float16* cp = xT + ((size_t)b * NPTS + n) * NCH + cb;
    const _Float16* np = xT + ((size_t)b * NPTS + m) * NCH + cb;
    const unsigned dC = (unsigned)(unsigned long long)(dstRow + cb);
    const unsigned dN = (unsigned)(unsigned long long)(dstRow + 64 + cb);
    // make sure all in-flight DS reads of the buffer being overwritten are done
    asm volatile("s_wait_dscnt 0x0" ::: "memory");
#pragma unroll
    for (int j = 0; j < 4; ++j) {
        async_ld_b128(dC + j * 16, cp + j * 8);
        async_ld_b128(dN + j * 16, np + j * 8);
    }
}

// ---------------------------------------------------------------------------
// One dense layer with register-resident weights:
//   H[64x16] = act(W[64xCIN] @ Y[CINx16] + bias) -> sY[:, chOut..chOut+63]
// B 32x16 layout: lane (lm,hi), col N=lm, K = kt*32 + hi*8 + [0..7] and +16.
// D: col N=lm, rows mt*16 + hi*8 + [0..7].
// ---------------------------------------------------------------------------
template<int KT, bool RELU>
__device__ __forceinline__ void dense_layer(const Frag wf[4][KT],
                                            const float bias[4][8],
                                            _Float16* sYw, int lm, int hi, int chOut)
{
    Frag bf[KT];                       // B fragments shared across all 4 M tiles
#pragma unroll
    for (int kt = 0; kt < KT; ++kt) {
        const _Float16* bp = sYw + lm * YSTR + kt * 32 + hi * 8;
        bf[kt].u[0] = *(const u4*)(bp);
        bf[kt].u[1] = *(const u4*)(bp + 16);
    }

    v8f acc[4];
#pragma unroll
    for (int mt = 0; mt < 4; ++mt) {
        acc[mt] = (v8f){0.f, 0.f, 0.f, 0.f, 0.f, 0.f, 0.f, 0.f};
#pragma unroll
        for (int kt = 0; kt < KT; ++kt) {
            acc[mt] = __builtin_amdgcn_wmma_f32_16x16x32_f16(
                false, wf[mt][kt].v, false, bf[kt].v, (short)0, acc[mt], false, false);
        }
    }

    // bias + activation + f16 pack + store (after ALL reads of this layer's input)
#pragma unroll
    for (int mt = 0; mt < 4; ++mt) {
        union { u4 u; _Float16 h[8]; } res;
#pragma unroll
        for (int g = 0; g < 8; ++g) {
            float v = acc[mt][g] + bias[mt][g];
            if (RELU) v = v > 0.f ? v : 0.f;
            res.h[g] = (_Float16)v;
        }
        *(u4*)(sYw + lm * YSTR + chOut + mt * 16 + hi * 8) = res.u;
    }
}

// ---------------------------------------------------------------------------
// Main kernel: 1 workgroup = 4 waves = 16 points (n0..n0+15). All weights and
// biases preloaded into ~550 VGPRs (wave32 has 1024), so the per-point loop is
// pure {async gather -> ds_load B frags -> 56 WMMA -> pack/store -> means}.
// ---------------------------------------------------------------------------
__global__ __launch_bounds__(128) void edgeconv_main(
    const float* __restrict__ x, const int* __restrict__ idx,
    const float* __restrict__ b0, const float* __restrict__ b1, const float* __restrict__ b2,
    const _Float16* __restrict__ xT,
    const _Float16* __restrict__ w0, const _Float16* __restrict__ w1, const _Float16* __restrict__ w2,
    float* __restrict__ out)
{
    __shared__ __align__(16) _Float16 sY[4][2][KNB][YSTR];  // double-buffered Y per wave
    __shared__ float sOut[192][16];                         // mean staging

    const int b    = blockIdx.y;
    const int n0   = blockIdx.x * 16;
    const int wv   = threadIdx.x >> 5;
    const int lane = threadIdx.x & 31;
    const int lm   = lane & 15;
    const int hi   = lane >> 4;
    const int k    = lm;        // neighbor row handled by this lane in the gather
    const int cb   = hi * 32;   // channel chunk handled by this lane in the gather

    // ---- preload all weights + biases into registers (immune to asm clobbers)
    Frag wf0[4][4], wf1[4][4], wf2[4][6];
    load_wfrags<128>(w0, lm, hi, wf0);
    load_wfrags<128>(w1, lm, hi, wf1);
    load_wfrags<192>(w2, lm, hi, wf2);
    float bs0[4][8], bs1[4][8], bs2[4][8];
#pragma unroll
    for (int mt = 0; mt < 4; ++mt) {
#pragma unroll
        for (int g = 0; g < 8; ++g) {
            const int ch = mt * 16 + hi * 8 + g;
            bs0[mt][g] = b0[ch];
            bs1[mt][g] = b1[ch];
            bs2[mt][g] = b2[ch];
        }
    }

    // prime the pipeline: point (it=0) -> buffer 0
    prefetch_point(xT, idx, b, n0 + wv, k, cb, &sY[wv][0][k][0]);

    for (int it = 0; it < 4; ++it) {
        const int p   = it * 4 + wv;
        const int cur = it & 1;

        if (it < 3) {
            prefetch_point(xT, idx, b, n0 + (it + 1) * 4 + wv, k, cb, &sY[wv][cur ^ 1][k][0]);
            wait_async<8>();    // 8 outstanding = next point's; current buffer complete
        } else {
            wait_async<0>();
        }

        _Float16* sYw = &sY[wv][cur][0][0];

        // ---- 3 conv layers via WMMA (W0 pre-folded for raw [center; knn]) ----
        dense_layer<4, true >(wf0, bs0, sYw, lm, hi, 64);    // H0 -> ch 64..127
        dense_layer<4, true >(wf1, bs1, sYw, lm, hi, 128);   // H1 -> ch 128..191
        dense_layer<6, false>(wf2, bs2, sYw, lm, hi, 192);   // H2 -> ch 192..255

        // ---- mean over K for channels 64..255 ----
#pragma unroll
        for (int j = 0; j < 6; ++j) {
            const int ch = lane + j * 32;   // 0..191 (Y channel = 64+ch)
            float s = 0.f;
#pragma unroll
            for (int kk = 0; kk < KNB; ++kk)
                s += (float)sYw[kk * YSTR + 64 + ch];
            sOut[ch][p] = s * 0.0625f;
        }
    }
    __syncthreads();

    // ---- coalesced output: out (B, 256, N); ch 0..63 = x exactly ----
    const int nOff = threadIdx.x & 15;
    const int chS  = threadIdx.x >> 4;   // 0..7
    for (int ch = chS; ch < 256; ch += 8) {
        float v;
        if (ch < 64) v = x[((size_t)b * NCH + ch) * NPTS + n0 + nOff];
        else         v = sOut[ch - 64][nOff];
        out[((size_t)b * 256 + ch) * NPTS + n0 + nOff] = v;
    }
}

// ---------------------------------------------------------------------------
extern "C" void kernel_launch(void* const* d_in, const int* in_sizes, int n_in,
                              void* d_out, int out_size, void* d_ws, size_t ws_size,
                              hipStream_t stream)
{
    const float* x   = (const float*)d_in[0];
    const int*   idx = (const int*)  d_in[1];
    const float* W0  = (const float*)d_in[2];
    const float* b0  = (const float*)d_in[3];
    const float* W1  = (const float*)d_in[4];
    const float* b1  = (const float*)d_in[5];
    const float* W2  = (const float*)d_in[6];
    const float* b2  = (const float*)d_in[7];
    float* out = (float*)d_out;

    // scratch layout: xT f16 (4 MB), then f16 weights
    _Float16* xT  = (_Float16*)d_ws;
    _Float16* w0h = xT  + (size_t)NB * NPTS * NCH;   // 2,097,152 halfs
    _Float16* w1h = w0h + 64 * 128;
    _Float16* w2h = w1h + 64 * 128;

    transpose_convert<<<dim3(NPTS / 64, NB), 256, 0, stream>>>(x, xT);
    convert_weights<<<dim3(112), 256, 0, stream>>>(W0, W1, W2, w0h, w1h, w2h);
    edgeconv_main<<<dim3(NPTS / 16, NB), 128, 0, stream>>>(
        x, idx, b0, b1, b2, xT, w0h, w1h, w2h, out);
}
